// DecoderBlock_30872224924345
// MI455X (gfx1250) — compile-verified, matrix operand order checked by reading, not run on hardware
//
#include <hip/hip_runtime.h>
#include <hip/hip_bf16.h>
#include <math.h>

#define T_SEQ   4096
#define C_DIM   768
#define F_DIM   3072
#define NHEADS  12
#define HD      64

typedef __attribute__((ext_vector_type(16))) __bf16        v16bf;
typedef __attribute__((ext_vector_type(8)))  float         v8f;
typedef __attribute__((ext_vector_type(8)))  unsigned int  v8u;
typedef unsigned short ushortT;

// round-to-nearest-even fp32 -> bf16 bits
__device__ __forceinline__ ushortT f2bf(float f) {
    unsigned int u = __builtin_bit_cast(unsigned int, f);
    u += 0x7fffu + ((u >> 16) & 1u);
    return (ushortT)(u >> 16);
}

// --- CDNA5 async global->LDS copy (ASYNCcnt) ------------------------------
// LDS address comes from the vdst VGPR; generic LDS pointer low 32 bits == wave-
// relative LDS offset, so truncation gives the right VGPR value.
__device__ __forceinline__ void async_ld_b128(void* lds, const void* gptr) {
    unsigned off = (unsigned)(unsigned long long)(uintptr_t)lds;
    unsigned long long ga = (unsigned long long)(uintptr_t)gptr;
    asm volatile("global_load_async_to_lds_b128 %0, %1, off"
                 :: "v"(off), "v"(ga) : "memory");
}
__device__ __forceinline__ void wait_async0() {
    asm volatile("s_wait_asynccnt 0" ::: "memory");
}

// Load a 16x32 bf16 fragment (A layout; B uses the same layout on an [N][K]
// tile) from a row-major LDS tile with row stride `ldh` halves (even).
// lanes 0-15: row M=lane, K={0..7,16..23}; lanes 16-31: row M=lane-16,
// K={8..15,24..31}.
__device__ __forceinline__ v16bf load_frag(const ushortT* base, int ldh) {
    const int lane = threadIdx.x & 31;
    const int row  = lane & 15;
    const int hi   = lane >> 4;
    const unsigned int* q0 = (const unsigned int*)(base + row * ldh + hi * 8);
    const unsigned int* q1 = (const unsigned int*)(base + row * ldh + hi * 8 + 16);
    v8u r;
    r[0] = q0[0]; r[1] = q0[1]; r[2] = q0[2]; r[3] = q0[3];
    r[4] = q1[0]; r[5] = q1[1]; r[6] = q1[2]; r[7] = q1[3];
    return __builtin_bit_cast(v16bf, r);
}

__device__ __forceinline__ float gelu_exact(float x) {
    return 0.5f * x * (1.0f + erff(x * 0.70710678118654752f));
}

// ---------------------------------------------------------------------------
// One-time converters
// ---------------------------------------------------------------------------
__global__ __launch_bounds__(256)
void cvt_f2bf_kernel(const float* __restrict__ in, ushortT* __restrict__ out, long n)
{
    long i = (long)blockIdx.x * 256 + threadIdx.x;
    long stride = (long)gridDim.x * 256;
    for (; i < n; i += stride) out[i] = f2bf(in[i]);
}

// in: fp32 [K][N]  ->  out: bf16 [N][K]
__global__ __launch_bounds__(256)
void transpose_f2bf_kernel(const float* __restrict__ in, ushortT* __restrict__ out,
                           int K, int N)
{
    __shared__ float tile[32][33];
    const int k0 = blockIdx.y * 32, n0 = blockIdx.x * 32;
    const int tx = threadIdx.x & 31, ty = threadIdx.x >> 5;   // 32 x 8
#pragma unroll
    for (int i = 0; i < 32; i += 8)
        tile[ty + i][tx] = in[(size_t)(k0 + ty + i) * N + n0 + tx];
    __syncthreads();
#pragma unroll
    for (int i = 0; i < 32; i += 8)
        out[(size_t)(n0 + ty + i) * K + k0 + tx] = f2bf(tile[tx][ty + i]);
}

// ---------------------------------------------------------------------------
// GEMM: C[M,N] = op( A[M,K] x B^T[N,K] + bias )
// A, Bt: bf16; outputs fp32 (Cout) and/or bf16 (Cbf), either may be null.
// op bit0: exact GELU ; op bit1: add fp32 residual res[M,N].
// Block tile 128x128, BK=32, 128 threads (4 waves), wave tile 64x64:
// 16 WMMAs per 16 ds_load_b128 per wave per k-step.
// Double-buffered LDS fed by async global->LDS B128 copies.
// ---------------------------------------------------------------------------
__global__ __launch_bounds__(128)
void gemm_bf16_wmma(const ushortT* __restrict__ A, const ushortT* __restrict__ Bt,
                    const float* __restrict__ bias, const float* __restrict__ res,
                    float* __restrict__ Cout, ushortT* __restrict__ Cbf,
                    int M, int N, int K, int op)
{
    // row stride 40 halves = 80B: async chunks and frag b128 loads are
    // 16B-aligned; banks r*20 mod 64 distinct over the 16 rows of a frag.
    __shared__ __align__(16) ushortT As[2][128][40];
    __shared__ __align__(16) ushortT Bs[2][128][40];

    const int t    = threadIdx.x;
    const int m0   = blockIdx.y * 128;
    const int n0   = blockIdx.x * 128;
    const int w    = t >> 5;
    const int wm   = w & 1;         // 2 wave-rows (64 rows each)
    const int wn   = w >> 1;        // 2 wave-cols (64 cols each)
    const int lane = t & 31, hi = lane >> 4, ln = lane & 15;
    const int nk   = K >> 5;

    auto issue = [&](int kt, int b) {
#pragma unroll
        for (int i = 0; i < 4; i++) {
            int c = t + i * 128;                  // 0..511
            int r = c >> 2, s2 = c & 3;           // 128 rows x 4 chunks of 16B
            async_ld_b128(&As[b][r][s2 * 8],
                          A  + (size_t)(m0 + r) * K + kt * 32 + s2 * 8);
            async_ld_b128(&Bs[b][r][s2 * 8],
                          Bt + (size_t)(n0 + r) * K + kt * 32 + s2 * 8);
        }
    };

    v8f acc[4][4] = {};
    issue(0, 0);

    for (int kt = 0; kt < nk; kt++) {
        wait_async0();
        __syncthreads();
        if (kt + 1 < nk) issue(kt + 1, (kt + 1) & 1);
        const int b = kt & 1;

        v16bf bf[4];
#pragma unroll
        for (int ni = 0; ni < 4; ni++)
            bf[ni] = load_frag(&Bs[b][wn * 64 + ni * 16][0], 40);
#pragma unroll
        for (int mi = 0; mi < 4; mi++) {
            v16bf af = load_frag(&As[b][wm * 64 + mi * 16][0], 40);
#pragma unroll
            for (int ni = 0; ni < 4; ni++)
                acc[mi][ni] = __builtin_amdgcn_wmma_f32_16x16x32_bf16(
                    false, af, false, bf[ni], (short)0, acc[mi][ni], false, false);
        }
    }

#pragma unroll
    for (int mi = 0; mi < 4; mi++)
#pragma unroll
        for (int ni = 0; ni < 4; ni++)
#pragma unroll
            for (int e = 0; e < 8; e++) {
                int row = m0 + wm * 64 + mi * 16 + e + hi * 8;
                int col = n0 + wn * 64 + ni * 16 + ln;
                float v = acc[mi][ni][e] + bias[col];
                if (op & 1) v = gelu_exact(v);
                if (op & 2) v += res[(size_t)row * N + col];
                if (Cout) Cout[(size_t)row * N + col] = v;
                if (Cbf)  Cbf [(size_t)row * N + col] = f2bf(v);
            }
}

// ---------------------------------------------------------------------------
// Causal flash attention on bf16 Q/K/V [T,C]; writes bf16 output.
// 128 threads = 4 waves; block = 64 query rows x 1 head.
// Q/K tiles arrive via async global->LDS copies; V is transposed via VGPRs.
// ---------------------------------------------------------------------------
__global__ __launch_bounds__(128)
void flash_attn_wmma(const ushortT* __restrict__ Q, const ushortT* __restrict__ Kx,
                     const ushortT* __restrict__ V, ushortT* __restrict__ O)
{
    // row stride 72 halves = 144B (16B aligned chunks, r*36 mod 64 distinct)
    __shared__ __align__(16) ushortT Qs[64][72];      // [q][d]
    __shared__ __align__(16) ushortT Ks[64][72];      // [s][d]  (Bt for Q*K^T)
    __shared__ __align__(16) ushortT Vt[64][72];      // [d][s]  (Bt for P*V)
    __shared__ __align__(16) ushortT Ps[4][16][72];   // per-wave P tile [q][s]

    const int t     = threadIdx.x;
    const int qbase = blockIdx.x * 64;
    const int head  = blockIdx.y;
    const int w     = t >> 5, lane = t & 31, hi = lane >> 4, ln = lane & 15;
    const size_t hoff = (size_t)head * HD;

#pragma unroll
    for (int i = 0; i < 4; i++) {                      // 64 rows x 8 chunks of 16B
        int c = t + i * 128;
        int r = c >> 3, s2 = c & 7;
        async_ld_b128(&Qs[r][s2 * 8], Q + (size_t)(qbase + r) * C_DIM + hoff + s2 * 8);
    }
    wait_async0();
    __syncthreads();

    v16bf aq[2];
    aq[0] = load_frag(&Qs[w * 16][0],  72);
    aq[1] = load_frag(&Qs[w * 16][32], 72);

    v8f o4[4] = {};
    float m_r[8], l_r[8];
#pragma unroll
    for (int e = 0; e < 8; e++) { m_r[e] = -1e30f; l_r[e] = 0.0f; }

    for (int kb = 0; kb <= qbase; kb += 64) {
        __syncthreads();                               // prior Ks/Vt/Ps reads done
#pragma unroll
        for (int i = 0; i < 4; i++) {
            int c = t + i * 128;
            int r = c >> 3, s2 = c & 7;
            async_ld_b128(&Ks[r][s2 * 8], Kx + (size_t)(kb + r) * C_DIM + hoff + s2 * 8);
        }
#pragma unroll
        for (int i = 0; i < 16; i++) {                 // V transpose via VGPRs
            int c = t + i * 128;
            int s = c >> 5, d = (c & 31) * 2;
            unsigned uv = *(const unsigned*)(V + (size_t)(kb + s) * C_DIM + hoff + d);
            Vt[d][s]     = (ushortT)uv;
            Vt[d + 1][s] = (ushortT)(uv >> 16);
        }
        wait_async0();
        __syncthreads();

        // S = Q * K^T (16x64 per wave)
        v8f s4[4] = {};
#pragma unroll
        for (int j = 0; j < 2; j++)
#pragma unroll
            for (int tt = 0; tt < 4; tt++) {
                v16bf bk = load_frag(&Ks[tt * 16][j * 32], 72);
                s4[tt] = __builtin_amdgcn_wmma_f32_16x16x32_bf16(
                    false, aq[j], false, bk, (short)0, s4[tt], false, false);
            }

        float sv[4][8];
#pragma unroll
        for (int tt = 0; tt < 4; tt++)
#pragma unroll
            for (int e = 0; e < 8; e++) {
                float vv = s4[tt][e] * 0.125f;         // 1/sqrt(64)
                int rg = qbase + w * 16 + e + hi * 8;
                int cg = kb + tt * 16 + ln;
                sv[tt][e] = (cg > rg) ? -1e30f : vv;
            }

        float rm[8];
#pragma unroll
        for (int e = 0; e < 8; e++)
            rm[e] = fmaxf(fmaxf(sv[0][e], sv[1][e]), fmaxf(sv[2][e], sv[3][e]));
#pragma unroll
        for (int off = 1; off < 16; off <<= 1)
#pragma unroll
            for (int e = 0; e < 8; e++)
                rm[e] = fmaxf(rm[e], __shfl_xor(rm[e], off, 32));

        float al[8];
#pragma unroll
        for (int e = 0; e < 8; e++) {
            float mn = fmaxf(m_r[e], rm[e]);
            al[e] = __expf(m_r[e] - mn);
            m_r[e] = mn;
        }

        float p[4][8], rs[8];
#pragma unroll
        for (int e = 0; e < 8; e++) rs[e] = 0.0f;
#pragma unroll
        for (int tt = 0; tt < 4; tt++)
#pragma unroll
            for (int e = 0; e < 8; e++) {
                p[tt][e] = __expf(sv[tt][e] - m_r[e]);
                rs[e] += p[tt][e];
            }
#pragma unroll
        for (int off = 1; off < 16; off <<= 1)
#pragma unroll
            for (int e = 0; e < 8; e++)
                rs[e] += __shfl_xor(rs[e], off, 32);
#pragma unroll
        for (int e = 0; e < 8; e++) l_r[e] = l_r[e] * al[e] + rs[e];
#pragma unroll
        for (int tt = 0; tt < 4; tt++)
#pragma unroll
            for (int e = 0; e < 8; e++) o4[tt][e] *= al[e];

        // stage P (bf16) through LDS to re-enter the A-fragment layout
#pragma unroll
        for (int tt = 0; tt < 4; tt++)
#pragma unroll
            for (int e = 0; e < 8; e++)
                Ps[w][e + hi * 8][tt * 16 + ln] = f2bf(p[tt][e]);
        __syncthreads();

        // O += P * V
#pragma unroll
        for (int j = 0; j < 2; j++) {
            v16bf ap = load_frag(&Ps[w][0][j * 32], 72);
#pragma unroll
            for (int tt = 0; tt < 4; tt++) {
                v16bf bv = load_frag(&Vt[tt * 16][j * 32], 72);
                o4[tt] = __builtin_amdgcn_wmma_f32_16x16x32_bf16(
                    false, ap, false, bv, (short)0, o4[tt], false, false);
            }
        }
    }

#pragma unroll
    for (int tt = 0; tt < 4; tt++)
#pragma unroll
        for (int e = 0; e < 8; e++) {
            int rg = qbase + w * 16 + e + hi * 8;
            int cg = tt * 16 + ln;
            O[(size_t)rg * C_DIM + hoff + cg] = f2bf(o4[tt][e] / l_r[e]);
        }
}

// ---------------------------------------------------------------------------
// LayerNorm: one block (256 threads) per row; fp32 out + optional bf16 out.
// ---------------------------------------------------------------------------
__global__ __launch_bounds__(256)
void layernorm_kernel(const float* __restrict__ in, const float* __restrict__ g,
                      const float* __restrict__ b, float* __restrict__ out,
                      ushortT* __restrict__ out_bf, int C)
{
    const int row = blockIdx.x;
    const float* p = in + (size_t)row * C;
    float s = 0.f, s2 = 0.f;
    for (int i = threadIdx.x; i < C; i += 256) { float v = p[i]; s += v; s2 += v * v; }
#pragma unroll
    for (int off = 1; off < 32; off <<= 1) {
        s  += __shfl_xor(s,  off, 32);
        s2 += __shfl_xor(s2, off, 32);
    }
    __shared__ float rs[8], rs2[8];
    if ((threadIdx.x & 31) == 0) { rs[threadIdx.x >> 5] = s; rs2[threadIdx.x >> 5] = s2; }
    __syncthreads();
    float ts = 0.f, ts2 = 0.f;
#pragma unroll
    for (int i = 0; i < 8; i++) { ts += rs[i]; ts2 += rs2[i]; }
    float mu  = ts / (float)C;
    float var = ts2 / (float)C - mu * mu;
    float inv = rsqrtf(var + 1e-5f);
    for (int i = threadIdx.x; i < C; i += 256) {
        float v = (p[i] - mu) * inv * g[i] + b[i];
        out[(size_t)row * C + i] = v;
        if (out_bf) out_bf[(size_t)row * C + i] = f2bf(v);
    }
}

// ---------------------------------------------------------------------------
extern "C" void kernel_launch(void* const* d_in, const int* in_sizes, int n_in,
                              void* d_out, int out_size, void* d_ws, size_t ws_size,
                              hipStream_t stream)
{
    const float* x   = (const float*)d_in[0];
    const float* Wq  = (const float*)d_in[1];
    const float* bq  = (const float*)d_in[2];
    const float* Wk  = (const float*)d_in[3];
    const float* bk  = (const float*)d_in[4];
    const float* Wv  = (const float*)d_in[5];
    const float* bv  = (const float*)d_in[6];
    const float* Wo  = (const float*)d_in[7];
    const float* bo  = (const float*)d_in[8];
    const float* g1  = (const float*)d_in[9];
    const float* bn1 = (const float*)d_in[10];
    const float* W1  = (const float*)d_in[11];
    const float* b1  = (const float*)d_in[12];
    const float* W2  = (const float*)d_in[13];
    const float* b2  = (const float*)d_in[14];
    const float* g2  = (const float*)d_in[15];
    const float* bn2 = (const float*)d_in[16];

    const size_t TC = (size_t)T_SEQ * C_DIM;
    const size_t TF = (size_t)T_SEQ * F_DIM;
    const size_t CC = (size_t)C_DIM * C_DIM;
    const size_t CF = (size_t)C_DIM * F_DIM;

    char* p = (char*)d_ws;
    auto alloc_bf = [&](size_t n) { ushortT* r = (ushortT*)p; p += ((n * 2 + 255) & ~(size_t)255); return r; };
    auto alloc_f  = [&](size_t n) { float*   r = (float*)p;   p += ((n * 4 + 255) & ~(size_t)255); return r; };

    ushortT* Xbf  = alloc_bf(TC);
    ushortT* WqT  = alloc_bf(CC);
    ushortT* WkT  = alloc_bf(CC);
    ushortT* WvT  = alloc_bf(CC);
    ushortT* WoT  = alloc_bf(CC);
    ushortT* W1T  = alloc_bf(CF);   // [F][C]
    ushortT* W2T  = alloc_bf(CF);   // [C][F]
    ushortT* Qb   = alloc_bf(TC);
    ushortT* Kb   = alloc_bf(TC);
    ushortT* Vb   = alloc_bf(TC);
    ushortT* Ab   = alloc_bf(TC);
    ushortT* Hbb  = alloc_bf(TC);
    ushortT* H1   = alloc_bf(TF);
    float*   S1   = alloc_f(TC);
    float*   Hb   = alloc_f(TC);
    float*   S2   = alloc_f(TC);

    dim3 cblk(256);
    dim3 gblk(128);
    dim3 gC(C_DIM / 128, T_SEQ / 128);
    dim3 gF(F_DIM / 128, T_SEQ / 128);

    // one-time precision conversion / weight transposition
    cvt_f2bf_kernel<<<dim3(1024), cblk, 0, stream>>>(x, Xbf, (long)TC);
    transpose_f2bf_kernel<<<dim3(C_DIM/32, C_DIM/32), cblk, 0, stream>>>(Wq, WqT, C_DIM, C_DIM);
    transpose_f2bf_kernel<<<dim3(C_DIM/32, C_DIM/32), cblk, 0, stream>>>(Wk, WkT, C_DIM, C_DIM);
    transpose_f2bf_kernel<<<dim3(C_DIM/32, C_DIM/32), cblk, 0, stream>>>(Wv, WvT, C_DIM, C_DIM);
    transpose_f2bf_kernel<<<dim3(C_DIM/32, C_DIM/32), cblk, 0, stream>>>(Wo, WoT, C_DIM, C_DIM);
    transpose_f2bf_kernel<<<dim3(F_DIM/32, C_DIM/32), cblk, 0, stream>>>(W1, W1T, C_DIM, F_DIM);
    transpose_f2bf_kernel<<<dim3(C_DIM/32, F_DIM/32), cblk, 0, stream>>>(W2, W2T, F_DIM, C_DIM);

    // QKV projections (bf16 outputs only)
    gemm_bf16_wmma<<<gC, gblk, 0, stream>>>(Xbf, WqT, bq, nullptr, nullptr, Qb, T_SEQ, C_DIM, C_DIM, 0);
    gemm_bf16_wmma<<<gC, gblk, 0, stream>>>(Xbf, WkT, bk, nullptr, nullptr, Kb, T_SEQ, C_DIM, C_DIM, 0);
    gemm_bf16_wmma<<<gC, gblk, 0, stream>>>(Xbf, WvT, bv, nullptr, nullptr, Vb, T_SEQ, C_DIM, C_DIM, 0);

    flash_attn_wmma<<<dim3(T_SEQ / 64, NHEADS), dim3(128), 0, stream>>>(Qb, Kb, Vb, Ab);

    // out-proj + residual(x), then LN1 (fp32 + bf16)
    gemm_bf16_wmma<<<gC, gblk, 0, stream>>>(Ab, WoT, bo, x, S1, nullptr, T_SEQ, C_DIM, C_DIM, 2);
    layernorm_kernel<<<dim3(T_SEQ), cblk, 0, stream>>>(S1, g1, bn1, Hb, Hbb, C_DIM);

    // MLP
    gemm_bf16_wmma<<<gF, gblk, 0, stream>>>(Hbb, W1T, b1, nullptr, nullptr, H1, T_SEQ, F_DIM, C_DIM, 1);
    gemm_bf16_wmma<<<gC, gblk, 0, stream>>>(H1, W2T, b2, Hb, S2, nullptr, T_SEQ, C_DIM, F_DIM, 2);
    layernorm_kernel<<<dim3(T_SEQ), cblk, 0, stream>>>(S2, g2, bn2, (float*)d_out, nullptr, C_DIM);
}
